// MPNN_25563645345834
// MI455X (gfx1250) — matrix operand order repaired and checked by reading, not compiled, not run on hardware
//
#include <hip/hip_runtime.h>
#include <stdint.h>

// ---------------------------------------------------------------------------
// MPNN (GCN x2 -> Dense x2 -> LSTM -> Dense) for MI455X / gfx1250, wave32.
// Heavy math on v_wmma_f32_16x16x32_bf16; a (33.5MB bf16) is L2-resident.
// Aggregation kernel: 32x64 tile/wave, register double-buffered K pipeline.
// ---------------------------------------------------------------------------

typedef __bf16  bf16_t;
typedef __bf16  v16bf __attribute__((ext_vector_type(16)));
typedef __bf16  v8bf  __attribute__((ext_vector_type(8)));
typedef float   v8f   __attribute__((ext_vector_type(8)));

#define NNODE 4096
#define TSTEP 32
#define NFEAT 64
#define NOUT  8

// --- WMMA fragment loaders (layouts per CDNA5 ISA 7.12.2, wave32) ----------
// A 16x32 bf16: lane m = lane&15; elements 0..7 -> K = kb..kb+7,
//               elements 8..15 -> K = kb+16..kb+23, kb = (lane>=16)*8.
__device__ __forceinline__ v16bf load_a_frag(const bf16_t* __restrict__ p0, int lda) {
    const int lane = threadIdx.x & 31;
    const int m  = lane & 15;
    const int kb = (lane >> 4) << 3;          // 0 or 8
    const bf16_t* p = p0 + (size_t)m * lda + kb;
    v8bf lo = *(const v8bf*)(p);              // K = kb .. kb+7   (16B load)
    v8bf hi = *(const v8bf*)(p + 16);         // K = kb+16 .. kb+23
    return __builtin_shufflevector(lo, hi, 0,1,2,3,4,5,6,7,8,9,10,11,12,13,14,15);
}

// B 32x16 bf16, source stored K-contiguous (column-major of B):
// B(k,n) at p0[n*ldb + k]; lane n = lane&15; elements 0..7 -> K = kb..kb+7,
// elements 8..15 -> K = kb+16..kb+23, kb = (lane>=16)*8. Two 16-byte loads.
__device__ __forceinline__ v16bf load_b_frag(const bf16_t* __restrict__ p0, int ldb) {
    const int lane = threadIdx.x & 31;
    const int n  = lane & 15;
    const int kb = (lane >> 4) << 3;          // 0 or 8
    const bf16_t* p = p0 + (size_t)n * ldb + kb;
    v8bf lo = *(const v8bf*)(p);
    v8bf hi = *(const v8bf*)(p + 16);
    return __builtin_shufflevector(lo, hi, 0,1,2,3,4,5,6,7,8,9,10,11,12,13,14,15);
}

__device__ __forceinline__ v8f wmma_bf16(v16bf a, v16bf b, v8f c) {
    return __builtin_amdgcn_wmma_f32_16x16x32_bf16(false, a, false, b, (short)0, c, false, false);
}

// --- casts -----------------------------------------------------------------
__global__ void k_cast_bf16(const float* __restrict__ src, bf16_t* __restrict__ dst, int count) {
    int i = blockIdx.x * blockDim.x + threadIdx.x;
    if (i < count) dst[i] = (bf16_t)src[i];
}

// src [K,C] f32 row-major -> dst [C,K] bf16 (K-contiguous for B fragments)
__global__ void k_castT_bf16(const float* __restrict__ src, bf16_t* __restrict__ dst, int K, int C) {
    int i = blockIdx.x * blockDim.x + threadIdx.x;
    if (i < K * C) {
        int c = i / K, k = i - c * K;
        dst[i] = (bf16_t)src[k * C + c];
    }
}

// --- feature GEMM:  out = A[T][N][64] @ W[64][64] (+bias), K=64 ------------
// mode 0: row-major bf16 out [T][N][64]; mode 1: transposed bf16 out [T][64][N]
__global__ __launch_bounds__(256) void k_gemm64(
    const bf16_t* __restrict__ A, const bf16_t* __restrict__ Wt,
    const float* __restrict__ bias,
    bf16_t* __restrict__ outRM, bf16_t* __restrict__ outT, int mode)
{
    const int t    = blockIdx.y;
    const int wave = threadIdx.x >> 5;
    const int lane = threadIdx.x & 31;
    const int row0 = (blockIdx.x * 8 + wave) * 16;     // grid.x=32 -> 256 row tiles
    const bf16_t* Abase = A + ((size_t)t * NNODE + row0) * NFEAT;
    v16bf a0 = load_a_frag(Abase +  0, NFEAT);
    v16bf a1 = load_a_frag(Abase + 32, NFEAT);
    v8f acc[4] = {};
#pragma unroll
    for (int ct = 0; ct < 4; ++ct) {
        const bf16_t* Wb = Wt + (size_t)(ct * 16) * NFEAT;
        acc[ct] = wmma_bf16(a0, load_b_frag(Wb +  0, NFEAT), acc[ct]);
        acc[ct] = wmma_bf16(a1, load_b_frag(Wb + 32, NFEAT), acc[ct]);
    }
    const int n  = lane & 15;
    const int mb = (lane >> 4) << 3;
    if (mode == 0) {
#pragma unroll
        for (int ct = 0; ct < 4; ++ct) {
            float bv = bias ? bias[ct * 16 + n] : 0.f;
#pragma unroll
            for (int r = 0; r < 8; ++r) {
                float v = acc[ct][r] + bv;
                outRM[((size_t)t * NNODE + row0 + mb + r) * NFEAT + ct * 16 + n] = (bf16_t)v;
            }
        }
    } else {
#pragma unroll
        for (int ct = 0; ct < 4; ++ct) {
            v8bf pack;
#pragma unroll
            for (int r = 0; r < 8; ++r) pack[r] = (bf16_t)acc[ct][r];
            *(v8bf*)(outT + ((size_t)t * NFEAT + ct * 16 + n) * NNODE + row0 + mb) = pack;
        }
    }
}

// --- aggregation:  Y[t] = relu(a @ XW_t + bias), K = 4096 ------------------
// aB [N][N] bf16 row-major; XWt [T][64][N] (K-contiguous); Y [T][N][64] bf16.
// Per wave: 32 rows x 64 cols (8 accumulators); K pipeline double-buffered in
// registers so WMMAs wait only on the previous stage's loads.
struct Frags { v16bf a0, a1, b0, b1, b2, b3; };

__device__ __forceinline__ Frags load_frags(const bf16_t* __restrict__ Arow0,
                                            const bf16_t* __restrict__ Arow1,
                                            const bf16_t* __restrict__ Bt, int k0) {
    Frags f;
    f.a0 = load_a_frag(Arow0 + k0, NNODE);
    f.a1 = load_a_frag(Arow1 + k0, NNODE);
    f.b0 = load_b_frag(Bt + (size_t)( 0) * NNODE + k0, NNODE);
    f.b1 = load_b_frag(Bt + (size_t)(16) * NNODE + k0, NNODE);
    f.b2 = load_b_frag(Bt + (size_t)(32) * NNODE + k0, NNODE);
    f.b3 = load_b_frag(Bt + (size_t)(48) * NNODE + k0, NNODE);
    return f;
}

__device__ __forceinline__ void frags_wmma(const Frags& f, v8f acc[2][4]) {
    acc[0][0] = wmma_bf16(f.a0, f.b0, acc[0][0]);
    acc[0][1] = wmma_bf16(f.a0, f.b1, acc[0][1]);
    acc[0][2] = wmma_bf16(f.a0, f.b2, acc[0][2]);
    acc[0][3] = wmma_bf16(f.a0, f.b3, acc[0][3]);
    acc[1][0] = wmma_bf16(f.a1, f.b0, acc[1][0]);
    acc[1][1] = wmma_bf16(f.a1, f.b1, acc[1][1]);
    acc[1][2] = wmma_bf16(f.a1, f.b2, acc[1][2]);
    acc[1][3] = wmma_bf16(f.a1, f.b3, acc[1][3]);
}

__global__ __launch_bounds__(256) void k_gcn_agg(
    const bf16_t* __restrict__ aB, const bf16_t* __restrict__ XWt,
    const float* __restrict__ bias, bf16_t* __restrict__ Y)
{
    const int t    = blockIdx.y;
    const int wave = threadIdx.x >> 5;
    const int lane = threadIdx.x & 31;
    const int row0 = (blockIdx.x * 8 + wave) * 32;     // grid.x=16 -> 128 wave tiles of 32 rows
    const bf16_t* Arow0 = aB + (size_t)row0 * NNODE;
    const bf16_t* Arow1 = aB + (size_t)(row0 + 16) * NNODE;
    const bf16_t* Bt    = XWt + (size_t)t * NFEAT * NNODE;

    v8f acc[2][4] = {};
    Frags cur = load_frags(Arow0, Arow1, Bt, 0);
    for (int k0 = 0; k0 < NNODE; k0 += 64) {
        __builtin_prefetch(Arow0 + k0 + 2048, 0, 1);   // global_prefetch of upcoming a rows
        Frags nxt = load_frags(Arow0, Arow1, Bt, k0 + 32);
        frags_wmma(cur, acc);                          // waits only on stage-older loads
        cur = load_frags(Arow0, Arow1, Bt, (k0 + 64) & (NNODE - 1)); // wraps on last iter
        frags_wmma(nxt, acc);
    }

    const int n  = lane & 15;
    const int mb = (lane >> 4) << 3;
#pragma unroll
    for (int rt = 0; rt < 2; ++rt) {
#pragma unroll
        for (int ct = 0; ct < 4; ++ct) {
            float bv = bias[ct * 16 + n];
#pragma unroll
            for (int r = 0; r < 8; ++r) {
                float v = acc[rt][ct][r] + bv;
                v = v > 0.f ? v : 0.f;
                Y[((size_t)t * NNODE + row0 + rt * 16 + mb + r) * NFEAT + ct * 16 + n] = (bf16_t)v;
            }
        }
    }
}

// --- Zx = SEQ @ wi + b_lstm  ->  f32 [N][T][32] (LSTM-friendly layout) -----
__global__ __launch_bounds__(256) void k_gemm_wi(
    const bf16_t* __restrict__ SEQ, const bf16_t* __restrict__ Wt,
    const float* __restrict__ bias, float* __restrict__ Zx)
{
    const int t    = blockIdx.y;
    const int wave = threadIdx.x >> 5;
    const int lane = threadIdx.x & 31;
    const int row0 = (blockIdx.x * 8 + wave) * 16;
    const bf16_t* Abase = SEQ + ((size_t)t * NNODE + row0) * NFEAT;
    v16bf a0 = load_a_frag(Abase +  0, NFEAT);
    v16bf a1 = load_a_frag(Abase + 32, NFEAT);
    v8f acc[2] = {};
#pragma unroll
    for (int ct = 0; ct < 2; ++ct) {
        const bf16_t* Wb = Wt + (size_t)(ct * 16) * NFEAT;
        acc[ct] = wmma_bf16(a0, load_b_frag(Wb +  0, NFEAT), acc[ct]);
        acc[ct] = wmma_bf16(a1, load_b_frag(Wb + 32, NFEAT), acc[ct]);
    }
    const int n  = lane & 15;
    const int mb = (lane >> 4) << 3;
#pragma unroll
    for (int ct = 0; ct < 2; ++ct) {
        float bv = bias[ct * 16 + n];
#pragma unroll
        for (int r = 0; r < 8; ++r) {
            int node = row0 + mb + r;
            Zx[(size_t)node * (TSTEP * 32) + t * 32 + ct * 16 + n] = acc[ct][r] + bv;
        }
    }
}

// --- sequential LSTM over N=4096 steps; batch=T=32, hidden=8 ---------------
// Zx[n][t][g] already holds x@wi + b; add h@wh recurrence in LDS.
__global__ __launch_bounds__(1024) void k_lstm(
    const float* __restrict__ Zx, const float* __restrict__ wh, float* __restrict__ hOut)
{
    __shared__ float hS[TSTEP][NOUT];
    __shared__ float zS[TSTEP][4 * NOUT];
    __shared__ float whS[NOUT][4 * NOUT];
    const int tid = threadIdx.x;
    const int t = tid >> 5, g = tid & 31;
    if (tid < NOUT * 4 * NOUT) whS[tid >> 5][tid & 31] = wh[tid];
    float c = 0.f;
    if (g < NOUT) hS[t][g] = 0.f;
    __syncthreads();
    for (int n = 0; n < NNODE; ++n) {
        float z = Zx[(size_t)n * (TSTEP * 32) + tid];  // coalesced: tid == t*32+g
#pragma unroll
        for (int j = 0; j < NOUT; ++j) z += hS[t][j] * whS[j][g];
        zS[t][g] = z;
        __syncthreads();
        if (g < NOUT) {
            float zi = zS[t][g], zf = zS[t][8 + g], zg = zS[t][16 + g], zo = zS[t][24 + g];
            float si = 1.f / (1.f + __expf(-zi));
            float sf = 1.f / (1.f + __expf(-zf));
            float so = 1.f / (1.f + __expf(-zo));
            c = sf * c + si * tanhf(zg);
            hS[t][g] = so * tanhf(c);
        }
        __syncthreads();
    }
    if (g < NOUT) hOut[t * NOUT + g] = hS[t][g];
}

// --- final dense: out[c] = b_d3[c] + sum_k h_flat[k] * w_d3[k][c] ----------
__global__ __launch_bounds__(256) void k_final(
    const float* __restrict__ hflat, const float* __restrict__ w,
    const float* __restrict__ b, float* __restrict__ out)
{
    __shared__ float hS[TSTEP * NOUT];
    const int tid = threadIdx.x;
    hS[tid] = hflat[tid];
    __syncthreads();
    const int c = blockIdx.x * 256 + tid;
    float acc = b[c];
#pragma unroll 4
    for (int k = 0; k < TSTEP * NOUT; ++k)
        acc += hS[k] * w[(size_t)k * (NOUT * NNODE) + c];
    out[c] = acc;
}

// ---------------------------------------------------------------------------
extern "C" void kernel_launch(void* const* d_in, const int* in_sizes, int n_in,
                              void* d_out, int out_size, void* d_ws, size_t ws_size,
                              hipStream_t stream) {
    (void)in_sizes; (void)n_in; (void)out_size; (void)ws_size;
    const float* graphs = (const float*)d_in[0];
    const float* a      = (const float*)d_in[1];
    const float* w_c1   = (const float*)d_in[2];
    const float* b_c1   = (const float*)d_in[3];
    const float* w_c2   = (const float*)d_in[4];
    const float* b_c2   = (const float*)d_in[5];
    const float* w_d1   = (const float*)d_in[6];
    const float* b_d1   = (const float*)d_in[7];
    const float* w_d2   = (const float*)d_in[8];
    const float* b_d2   = (const float*)d_in[9];
    const float* wi     = (const float*)d_in[10];
    const float* wh     = (const float*)d_in[11];
    const float* b_lstm = (const float*)d_in[12];
    const float* w_d3   = (const float*)d_in[13];
    const float* b_d3   = (const float*)d_in[14];
    float* out = (float*)d_out;

    // workspace layout (~84 MB, all offsets 256B-aligned)
    uint8_t* ws = (uint8_t*)d_ws;
    size_t off = 0;
    bf16_t* aB    = (bf16_t*)(ws + off); off += (size_t)NNODE * NNODE * 2;          // 33.5 MB
    bf16_t* bufT  = (bf16_t*)(ws + off); off += (size_t)TSTEP * NFEAT * NNODE * 2;  // 16.8 MB
    bf16_t* bufR1 = (bf16_t*)(ws + off); off += (size_t)TSTEP * NNODE * NFEAT * 2;  // 16.8 MB
    bf16_t* bufR2 = (bf16_t*)(ws + off); off += (size_t)TSTEP * NNODE * NFEAT * 2;  // 16.8 MB
    bf16_t* wc1t  = (bf16_t*)(ws + off); off += 4096 * 2;
    bf16_t* wc2t  = (bf16_t*)(ws + off); off += 4096 * 2;
    bf16_t* wd1t  = (bf16_t*)(ws + off); off += 4096 * 2;
    bf16_t* wd2t  = (bf16_t*)(ws + off); off += 4096 * 2;
    bf16_t* wit   = (bf16_t*)(ws + off); off += 2048 * 2;
    float*  hOut  = (float*)(ws + off);  off += 256 * 4;
    float*  Zx    = (float*)bufT;  // reuse: XW buffer dead before Zx is written (same 16.8 MB)

    // 1) casts to bf16 (a is the big one; L2-resident afterwards)
    const int cA = NNODE * NNODE;
    k_cast_bf16<<<(cA + 255) / 256, 256, 0, stream>>>(a, aB, cA);
    const int cG = TSTEP * NNODE * NFEAT;
    k_cast_bf16<<<(cG + 255) / 256, 256, 0, stream>>>(graphs, bufR1, cG);
    k_castT_bf16<<<16, 256, 0, stream>>>(w_c1, wc1t, 64, 64);
    k_castT_bf16<<<16, 256, 0, stream>>>(w_c2, wc2t, 64, 64);
    k_castT_bf16<<<16, 256, 0, stream>>>(w_d1, wd1t, 64, 64);
    k_castT_bf16<<<16, 256, 0, stream>>>(w_d2, wd2t, 64, 64);
    k_castT_bf16<<< 8, 256, 0, stream>>>(wi,   wit,  64, 32);

    const dim3 gF(32, 32);  // feature GEMMs: 256 row tiles of 16
    const dim3 gA(16, 32);  // aggregation:   128 wave tiles of 32 rows
    // 2) GCN layer 1: XW1 = x @ w_c1 (transposed), Y1 = relu(a @ XW1 + b_c1)
    k_gemm64 <<<gF, 256, 0, stream>>>(bufR1, wc1t, nullptr, nullptr, bufT, 1);
    k_gcn_agg<<<gA, 256, 0, stream>>>(aB, bufT, b_c1, bufR2);
    // 3) GCN layer 2
    k_gemm64 <<<gF, 256, 0, stream>>>(bufR2, wc2t, nullptr, nullptr, bufT, 1);
    k_gcn_agg<<<gA, 256, 0, stream>>>(aB, bufT, b_c2, bufR1);
    // 4) Dense d1, d2 (bias, no relu)
    k_gemm64 <<<gF, 256, 0, stream>>>(bufR1, wd1t, b_d1, bufR2, nullptr, 0);
    k_gemm64 <<<gF, 256, 0, stream>>>(bufR2, wd2t, b_d2, bufR1, nullptr, 0);
    // 5) Zx = seq @ wi + b_lstm  (f32, [N][T][32])
    k_gemm_wi<<<gF, 256, 0, stream>>>(bufR1, wit, b_lstm, Zx);
    // 6) sequential LSTM (one 1024-thread workgroup)
    k_lstm<<<1, 1024, 0, stream>>>(Zx, wh, hOut);
    // 7) final dense -> d_out [N*H]
    k_final<<<128, 256, 0, stream>>>(hOut, w_d3, b_d3, out);
}
